// VectorQuantizer_56581899157657
// MI455X (gfx1250) — matrix operand order repaired and checked by reading, not compiled, not run on hardware
//
#include <hip/hip_runtime.h>
#include <hip/hip_bf16.h>

typedef _Float16 v8h  __attribute__((ext_vector_type(8)));
typedef _Float16 v16h __attribute__((ext_vector_type(16)));
typedef float    v8f  __attribute__((ext_vector_type(8)));

#define N_ROWS   32768      // B*H*W = 32*32*32
#define N_CODES  8192
#define CDIM     256
#define HW       1024       // H*W
#define M_TILE   64
#define N_STEP   64
#define NS_TOT   (N_CODES / N_STEP)   // 128
#define SCALE    512.0f
#define ZSTRIDE  264        // padded stride for the one-time z stage (halves)
#define OUT_ZQ_N 8388608    // 32*256*32*32
#define N_ELEMS  8388608

// workspace layout (units: floats)
#define WS_PROJT  0
#define WS_CB32   (WS_PROJT + 256*256)                // 65536
#define WS_CB16   (WS_CB32 + N_CODES*CDIM)            // f16 codebook lives here
#define WS_NORM   (WS_CB16 + (N_CODES*CDIM)/2)
#define WS_ACCUM  (WS_NORM + N_CODES)

typedef __attribute__((address_space(3))) const void* lds_cptr;
static __device__ __forceinline__ unsigned ldsOffset(const void* p) {
    return (unsigned)(unsigned long long)(lds_cptr)p;
}

// async copy of 16B global -> LDS (ISA 15.18.3 op 98), tracked by ASYNCcnt
#define ASYNC_LOAD_B128(ldsoff, voff, sbase)                        \
    asm volatile("global_load_async_to_lds_b128 %0, %1, %2"         \
                 :: "v"(ldsoff), "v"(voff), "s"(sbase) : "memory")
#define WAIT_ASYNCCNT(n) asm volatile("s_wait_asynccnt " #n ::: "memory")

// ---------------------------------------------------------------- kernel 1
// projT[k][c] = proj[c][k]; zero the loss accumulator.
__global__ void vq_prep(const float* __restrict__ proj, float* __restrict__ ws) {
    int k = blockIdx.x;      // 0..255
    int c = threadIdx.x;     // 0..255
    ws[WS_PROJT + k * 256 + c] = proj[c * 256 + k];
    if (k == 0 && c == 0) ws[WS_ACCUM] = 0.0f;
}

// ---------------------------------------------------------------- kernel 2
// codebook = emb @ projT (fp32 exact). Emit fp32 codebook, scaled-f16
// codebook, and norms of the f16-rounded scaled rows.
__global__ __launch_bounds__(256) void vq_codebook(const float* __restrict__ emb,
                                                   float* __restrict__ ws) {
    __shared__ float embL[8 * CDIM];
    __shared__ float red[256];
    const float* projT = ws + WS_PROJT;
    float*       cb32  = ws + WS_CB32;
    _Float16*    cb16  = (_Float16*)(ws + WS_CB16);
    float*       norms = ws + WS_NORM;

    int n0 = blockIdx.x * 8;
    int t  = threadIdx.x;

    #pragma unroll
    for (int j = 0; j < 8; ++j)
        embL[j * CDIM + t] = emb[(size_t)(n0 + j) * CDIM + t];
    __syncthreads();

    float acc[8];
    #pragma unroll
    for (int j = 0; j < 8; ++j) acc[j] = 0.0f;

    for (int k = 0; k < CDIM; ++k) {
        float p = projT[k * 256 + t];          // coalesced, L2-resident
        #pragma unroll
        for (int j = 0; j < 8; ++j) acc[j] = fmaf(embL[j * CDIM + k], p, acc[j]);
    }

    for (int j = 0; j < 8; ++j) {
        float v  = acc[j];
        float vs = v * SCALE;
        _Float16 vh = (_Float16)vs;
        cb32[(size_t)(n0 + j) * CDIM + t] = v;
        cb16[(size_t)(n0 + j) * CDIM + t] = vh;
        float vr = (float)vh;                  // norm of the rounded value
        red[t] = vr * vr;
        __syncthreads();
        for (int s = 128; s > 0; s >>= 1) {
            if (t < s) red[t] += red[t + s];
            __syncthreads();
        }
        if (t == 0) norms[n0 + j] = red[0];
        __syncthreads();
    }
}

// async-stage one 64x256-half codebook tile into buf[ns&1], XOR-16B swizzled
static __device__ __forceinline__ void stage_tile(unsigned smemBase,
                                                  const _Float16* cb16,
                                                  int ns, int row, int seg) {
    unsigned long long sbase =
        (unsigned long long)(const void*)(cb16 + (size_t)(ns * N_STEP) * CDIM);
    unsigned rbase = smemBase + ((unsigned)(ns & 1) << 15) + (unsigned)(row * 512);
    unsigned gbase = (unsigned)(row * 512);
    #pragma unroll
    for (int q = 0; q < 8; ++q) {
        int chunk = seg * 8 + q;                            // 0..31 (16B units)
        unsigned lds  = rbase + (unsigned)(((chunk ^ (row & 15)) << 4));
        unsigned voff = gbase + (unsigned)(chunk << 4);
        ASYNC_LOAD_B128(lds, voff, sbase);
    }
}

// ---------------------------------------------------------------- kernel 3
// Fused distance + argmin + gather + loss.  Block = 64 z-rows, 8 waves.
// A fragments live in registers for the whole sweep; 64KB LDS = two 32KB
// swizzled codebook tile buffers (double-buffered async fills), aliased with
// the one-time z stage and the argmin-reduction scratch.
__global__ __launch_bounds__(256) void vq_main(const float* __restrict__ z,
                                               float* __restrict__ ws,
                                               float* __restrict__ out) {
    __shared__ __align__(16) unsigned char smem[65536];

    _Float16* zA   = (_Float16*)smem;             // phase 1 (stride ZSTRIDE)
    float*    redV = (float*)smem;                // phase 3: [64][64] floats
    int*      redI = (int*)(smem + 16384);        // phase 3: [64][64] ints
    int*      bestRowIdx = (int*)(smem + 32768);  // phase 3/4 (buf1 region)

    const _Float16* cb16  = (const _Float16*)(ws + WS_CB16);
    const float*    cb32  = ws + WS_CB32;
    const float*    norms = ws + WS_NORM;

    int t   = threadIdx.x;
    int R0  = blockIdx.x * M_TILE;                // global z-row base
    int b   = R0 / HW;
    int hw0 = R0 % HW;

    // ---- phase 1a: stage z tile to LDS as f16 (transposed, padded) ----
    {
        int hwl = t & 63;
        int cg0 = (t >> 6) * 64;
        for (int i = 0; i < 64; ++i) {
            int c = cg0 + i;
            float zv = z[((size_t)(b * CDIM + c)) * HW + hw0 + hwl];
            zA[hwl * ZSTRIDE + c] = (_Float16)zv;
        }
    }
    __syncthreads();

    int lane = t & 31;
    int lh   = lane & 15;
    int hi   = lane >> 4;
    int w    = t >> 5;                 // wave id 0..7
    int mp   = w >> 2;                 // M pair: rows [mp*32, mp*32+32)
    int nt   = w & 3;                  // N tile:  cols nt*16 within the 64-step

    // ---- phase 1b: pull all A fragments into registers (2 x 8 x v16h) ----
    union Frag { v16h v; v8h p[2]; };
    Frag Af[2][8];
    #pragma unroll
    for (int a = 0; a < 2; ++a) {
        #pragma unroll
        for (int ks = 0; ks < 8; ++ks) {
            // A 16x32 f16 layout: lanes<16 hold K 0-7 & 16-23, lanes>=16 hold 8-15 & 24-31
            const _Float16* ar = zA + (mp * 32 + a * 16 + lh) * ZSTRIDE + ks * 32 + hi * 8;
            Af[a][ks].p[0] = *(const v8h*)ar;
            Af[a][ks].p[1] = *(const v8h*)(ar + 16);
        }
    }
    __syncthreads();                   // zA fully consumed; LDS now belongs to cb bufs

    float bestVal[2][8];
    int   bestIdx[2][8];
    #pragma unroll
    for (int a = 0; a < 2; ++a)
        #pragma unroll
        for (int r = 0; r < 8; ++r) { bestVal[a][r] = 3.4e38f; bestIdx[a][r] = 0; }

    unsigned smemBase = ldsOffset(smem);
    int      row = t >> 2;             // 0..63  (staging role)
    int      seg = t & 3;              // 0..3   (128B chunk group within a row)

    stage_tile(smemBase, cb16, 0, row, seg);   // prologue fill of buf0

    // ---- phase 2: sweep all 8192 codes, double-buffered, 64/step ----
    for (int ns = 0; ns < NS_TOT; ++ns) {
        __syncthreads();               // everyone done with buf[(ns+1)&1]'s old data
        if (ns + 1 < NS_TOT) {
            stage_tile(smemBase, cb16, ns + 1, row, seg);
            WAIT_ASYNCCNT(0x8);        // in-order: buf[ns&1]'s 8 loads complete
        } else {
            WAIT_ASYNCCNT(0x0);
        }
        __syncthreads();               // tile ns visible to all waves

        int n0 = ns * N_STEP;
        float nrm = norms[n0 + nt * 16 + lh];
        const unsigned char* rowPtr =
            smem + ((unsigned)(ns & 1) << 15) + (unsigned)((nt * 16 + lh) * 512);

        v8f acc0 = {};
        v8f acc1 = {};
        #pragma unroll
        for (int ks = 0; ks < 8; ++ks) {
            // B 32x16 f16: lanes<16 hold K 0-7 & 16-23 of col N=lane, lanes>=16 hold K 8-15 & 24-31
            int c0 = ks * 4 + hi;      // 16B-chunk index within the row
            union Frag Bf;
            Bf.p[0] = *(const v8h*)(rowPtr + (((c0    ) ^ lh) << 4));
            Bf.p[1] = *(const v8h*)(rowPtr + (((c0 + 2) ^ lh) << 4));
            acc0 = __builtin_amdgcn_wmma_f32_16x16x32_f16(
                       false, Af[0][ks].v, false, Bf.v, (short)0, acc0, false, false);
            acc1 = __builtin_amdgcn_wmma_f32_16x16x32_f16(
                       false, Af[1][ks].v, false, Bf.v, (short)0, acc1, false, false);
        }

        int nIdx = n0 + nt * 16 + lh;
        #pragma unroll
        for (int r = 0; r < 8; ++r) {
            float s0 = nrm - 2.0f * acc0[r];   // S^2*||e||^2 - 2S*(z.e)
            float s1 = nrm - 2.0f * acc1[r];
            if (s0 < bestVal[0][r]) { bestVal[0][r] = s0; bestIdx[0][r] = nIdx; }
            if (s1 < bestVal[1][r]) { bestVal[1][r] = s1; bestIdx[1][r] = nIdx; }
        }
    }

    // ---- phase 3: cross-lane / cross-wave argmin (reuse LDS region) ----
    __syncthreads();
    #pragma unroll
    for (int a = 0; a < 2; ++a) {
        #pragma unroll
        for (int r = 0; r < 8; ++r) {
            int m    = mp * 32 + a * 16 + r + hi * 8; // C layout: vgpr r -> rows r, r+8
            int slot = nt * 16 + lh;
            redV[m * 64 + slot] = bestVal[a][r];
            redI[m * 64 + slot] = bestIdx[a][r];
        }
    }
    __syncthreads();
    if (t < M_TILE) {
        float bv = redV[t * 64];
        int   bi = redI[t * 64];
        for (int s2 = 1; s2 < 64; ++s2) {
            float v2 = redV[t * 64 + s2];
            int   i2 = redI[t * 64 + s2];
            if (v2 < bv || (v2 == bv && i2 < bi)) { bv = v2; bi = i2; }
        }
        bestRowIdx[t] = bi;
        out[OUT_ZQ_N + 3 + R0 + t] = (float)bi;    // min_encoding_indices
    }
    __syncthreads();

    // ---- phase 4: gather fp32 codebook rows, write zq, fused loss ----
    float lsum = 0.0f;
    {
        int hwl = t & 63;
        int cg0 = (t >> 6) * 64;
        int idx = bestRowIdx[hwl];
        for (int i = 0; i < 64; ++i) {
            int c = cg0 + i;
            float q = cb32[(size_t)idx * CDIM + c];
            size_t gi = ((size_t)(b * CDIM + c)) * HW + hw0 + hwl;
            float d = q - z[gi];
            lsum = fmaf(d, d, lsum);
            out[gi] = q;                           // straight-through value == zq
        }
    }
    __syncthreads();
    redV[t] = lsum;
    __syncthreads();
    for (int s = 128; s > 0; s >>= 1) {
        if (t < s) redV[t] += redV[t + s];
        __syncthreads();
    }
    if (t == 0) atomicAdd(ws + WS_ACCUM, redV[0]);
}

// ---------------------------------------------------------------- kernel 4
__global__ void vq_finalize(const float* __restrict__ ws, float* __restrict__ out) {
    float mse = ws[WS_ACCUM] * (1.0f / (float)N_ELEMS);
    out[OUT_ZQ_N + 0] = 1.25f * mse;   // loss
    out[OUT_ZQ_N + 1] = 0.25f * mse;   // commitment_loss
    out[OUT_ZQ_N + 2] = mse;           // codebook_loss
}

extern "C" void kernel_launch(void* const* d_in, const int* in_sizes, int n_in,
                              void* d_out, int out_size, void* d_ws, size_t ws_size,
                              hipStream_t stream) {
    const float* z    = (const float*)d_in[0];   // [32,256,32,32]
    const float* emb  = (const float*)d_in[1];   // [8192,256]
    const float* proj = (const float*)d_in[2];   // [256,256]
    float* ws  = (float*)d_ws;
    float* out = (float*)d_out;

    vq_prep    <<<256,               256, 0, stream>>>(proj, ws);
    vq_codebook<<<N_CODES / 8,       256, 0, stream>>>(emb, ws);
    vq_main    <<<N_ROWS / M_TILE,   256, 0, stream>>>(z, ws, out);
    vq_finalize<<<1,                 1,   0, stream>>>(ws, out);
}
